// PWD_Module_YOLOv5_14877766713777
// MI455X (gfx1250) — compile-verified, matrix operand order checked by reading, not run on hardware
//
#include <hip/hip_runtime.h>
#include <hip/hip_bf16.h>
#include <math.h>

typedef _Float16 v8h  __attribute__((ext_vector_type(8)));
typedef _Float16 v16h __attribute__((ext_vector_type(16)));
typedef float    v8f  __attribute__((ext_vector_type(8)));
typedef int      v4i  __attribute__((ext_vector_type(4)));

// ---- gfx1250 async global->LDS path (ASYNCcnt), with synchronous fallback --
#if defined(__has_builtin)
#  if __has_builtin(__builtin_amdgcn_global_load_async_to_lds_b128)
#    define HAVE_ASYNC_LDS 1
#  endif
#endif
#ifndef HAVE_ASYNC_LDS
#  define HAVE_ASYNC_LDS 0
#endif

__device__ __forceinline__ void async_wait0() {
#if defined(__has_builtin) && __has_builtin(__builtin_amdgcn_s_wait_asynccnt)
    __builtin_amdgcn_s_wait_asynccnt(0);
#else
    asm volatile("s_wait_asynccnt 0x0" ::: "memory");
#endif
}

#if HAVE_ASYNC_LDS
// b128 async copy: global -> LDS, no VGPR round-trip, tracked by ASYNCcnt.
// Param 0 is a global-AS int4* per the compiler's own diagnostic.
__device__ __forceinline__ void async_copy16(const _Float16* gsrc, _Float16* ldst) {
    __builtin_amdgcn_global_load_async_to_lds_b128(
        (__attribute__((address_space(1))) v4i*)gsrc,
        (__attribute__((address_space(3))) v4i*)ldst,
        0, 0);
}
#endif

// ---------------------------------------------------------------------------
// Zero the two scalar accumulators in workspace.
// ---------------------------------------------------------------------------
__global__ void init_acc_kernel(float* __restrict__ acc) {
    if (blockIdx.x == 0 && threadIdx.x < 2) acc[threadIdx.x] = 0.0f;
}

// ---------------------------------------------------------------------------
// Per-pixel channel-L2 normalization + transpose [b,c,hw] -> [b,hw,c] in f16.
// eps added AFTER sqrt (matches reference).
// ---------------------------------------------------------------------------
__global__ __launch_bounds__(256) void normalize_kernel(
    const float* __restrict__ in, _Float16* __restrict__ out,
    int c, int hw, int total)
{
    int idx = blockIdx.x * blockDim.x + threadIdx.x;
    if (idx >= total) return;
    int bb  = idx / hw;
    int pix = idx - bb * hw;
    const float* src = in + (size_t)bb * c * hw + pix;
    float ss = 0.0f;
    for (int ch = 0; ch < c; ++ch) {
        float v = src[(size_t)ch * hw];
        ss += v * v;
    }
    float inv = 1.0f / (sqrtf(ss) + 1e-8f);
    _Float16* dst = out + ((size_t)bb * hw + pix) * c;
    for (int ch = 0; ch < c; ++ch)
        dst[ch] = (_Float16)(src[(size_t)ch * hw] * inv);
}

// ---------------------------------------------------------------------------
// Assemble a 16x32 f16 WMMA fragment slice for this lane from two 16B chunks.
// 16-bit A/B layout: lanes 0-15 hold K in [0,8)+[16,24), lanes 16-31 hold
// K in [8,16)+[24,32)  => two 16-byte contiguous loads (global or LDS).
// ---------------------------------------------------------------------------
__device__ __forceinline__ v16h load_frag(const _Float16* p) {
    v8h lo = *(const v8h*)(p);
    v8h hi = *(const v8h*)(p + 16);
    return __builtin_shufflevector(lo, hi, 0,1,2,3,4,5,6,7,8,9,10,11,12,13,14,15);
}

#define WMMA_F16(A, B, C) \
    __builtin_amdgcn_wmma_f32_16x16x32_f16(false, (A), false, (B), (short)0, (C), false, false)

// ---------------------------------------------------------------------------
// LDS-staged, symmetry-aware Gram-difference kernel.
//
// One 256-thread block (8 waves, 4x2 wave grid) computes a 128x64 macro-tile
// of both Gram matrices. Macro-tiles are enumerated only on/above the
// diagonal; per-32x32-tile weights (2 above diag, 1 on diag, 0 below)
// recover the full symmetric sum.
//
// Staging is branch-free: the 384-row x 64B k-slice panel is 1536 16B
// chunks = 256 threads x 6; pointer pairs hoisted out of the k-loop.
// With HAVE_ASYNC_LDS the chunks go memory->LDS via
// global_load_async_to_lds_b128 (no VGPR round-trip, one s_wait_asynccnt).
// ---------------------------------------------------------------------------
#define LDS_ROW 40   // 32 f16 payload + 8 f16 pad (80B, 16B aligned)

template <int C>
__global__ __launch_bounds__(256) void gram_diff_lds_kernel(
    const _Float16* __restrict__ Pt, const _Float16* __restrict__ Ps,
    float* __restrict__ acc, int hw, int n64)
{
    __shared__ __align__(16) _Float16 lds[384 * LDS_ROW];

    const int tid  = threadIdx.x;
    const int lane = tid & 31;
    const int wave = tid >> 5;
    const int wm   = wave & 3;   // wave row-tile 0..3
    const int wn   = wave >> 2;  // wave col-tile 0..1
    const int bb   = blockIdx.y;

    // ---- triangle macro-tile decode: f(t) = t*(n64+1-t), TN = 2*TM + rem --
    const int idx   = (int)blockIdx.x;
    const int nHalf = n64 >> 1;
    float nn  = (float)(n64 + 1);
    float dsc = nn * nn - 4.0f * (float)idx;
    int tmb = (int)((nn - sqrtf(fmaxf(dsc, 0.0f))) * 0.5f);
    if (tmb < 0) tmb = 0;
    if (tmb > nHalf - 1) tmb = nHalf - 1;
    while (tmb > 0 && tmb * (n64 + 1 - tmb) > idx) --tmb;
    while (tmb < nHalf - 1 && (tmb + 1) * (n64 - tmb) <= idx) ++tmb;
    const int tnb = 2 * tmb + (idx - tmb * (n64 + 1 - tmb));

    const int M0 = tmb * 128;
    const int N0 = tnb * 64;

    const _Float16* Ptb = Pt + (size_t)bb * hw * C;
    const _Float16* Psb = Ps + (size_t)bb * hw * C;

    // symmetry weight for this wave's 32x32 tile
    const int tmw = tmb * 4 + wm;
    const int tnw = tnb * 2 + wn;
    const float wgt = (tnw > tmw) ? 2.0f : ((tnw == tmw) ? 1.0f : 0.0f);

    // ---- hoisted branch-free staging pointers: 6 chunks of 16B/thread ----
    const _Float16* srcp[6];
    _Float16*       dstp[6];
#pragma unroll
    for (int i = 0; i < 6; ++i) {
        int chunk = tid + i * 256;       // 0..1535
        int r     = chunk >> 2;          // panel row 0..383
        int part  = (chunk & 3) * 8;     // f16 offset within 32-f16 row slice
        const _Float16* s;
        if      (r < 128) s = Ptb + (size_t)(M0 + r)       * C;
        else if (r < 256) s = Psb + (size_t)(M0 + r - 128) * C;
        else if (r < 320) s = Ptb + (size_t)(N0 + r - 256) * C;
        else              s = Psb + (size_t)(N0 + r - 320) * C;
        srcp[i] = s + part;
        dstp[i] = &lds[r * LDS_ROW + part];
    }

    // per-lane fragment pointers into LDS (constant across k-steps)
    const int rm  = lane & 15;
    const int hi8 = (lane >> 4) << 3;
    const _Float16* aT0p = &lds[(      wm * 32 + rm) * LDS_ROW + hi8];
    const _Float16* aT1p = aT0p + 16 * LDS_ROW;
    const _Float16* aS0p = &lds[(128 + wm * 32 + rm) * LDS_ROW + hi8];
    const _Float16* aS1p = aS0p + 16 * LDS_ROW;
    const _Float16* bT0p = &lds[(256 + wn * 32 + rm) * LDS_ROW + hi8];
    const _Float16* bT1p = bT0p + 16 * LDS_ROW;
    const _Float16* bS0p = &lds[(320 + wn * 32 + rm) * LDS_ROW + hi8];
    const _Float16* bS1p = bS0p + 16 * LDS_ROW;

    v8f ct00 = {}, ct01 = {}, ct10 = {}, ct11 = {};
    v8f cs00 = {}, cs01 = {}, cs10 = {}, cs11 = {};

#pragma unroll
    for (int kk = 0; kk < C; kk += 32) {
        __syncthreads();  // previous k-step's fragment reads complete

        // ---- branch-free cooperative staging (kk folds into load offset) --
#if HAVE_ASYNC_LDS
#pragma unroll
        for (int i = 0; i < 6; ++i)
            async_copy16(srcp[i] + kk, dstp[i]);
        async_wait0();    // all 6 chunks resident in LDS
#else
#pragma unroll
        for (int i = 0; i < 6; ++i)
            *(v8h*)dstp[i] = *(const v8h*)(srcp[i] + kk);
#endif

        __syncthreads();  // panel visible to all waves

        // ---- fragment loads from LDS + 8 WMMAs ---------------------------
        v16h a0 = load_frag(aT0p);
        v16h a1 = load_frag(aT1p);
        v16h b0 = load_frag(bT0p);
        v16h b1 = load_frag(bT1p);
        ct00 = WMMA_F16(a0, b0, ct00);
        ct01 = WMMA_F16(a0, b1, ct01);
        ct10 = WMMA_F16(a1, b0, ct10);
        ct11 = WMMA_F16(a1, b1, ct11);

        v16h c0 = load_frag(aS0p);
        v16h c1 = load_frag(aS1p);
        v16h d0 = load_frag(bS0p);
        v16h d1 = load_frag(bS1p);
        cs00 = WMMA_F16(c0, d0, cs00);
        cs01 = WMMA_F16(c0, d1, cs01);
        cs10 = WMMA_F16(c1, d0, cs10);
        cs11 = WMMA_F16(c1, d1, cs11);
    }

    // ---- (St - Ss)^2, symmetry weight, wave32 butterfly reduce -----------
    float partial = 0.0f;
#pragma unroll
    for (int e = 0; e < 8; ++e) {
        float d;
        d = ct00[e] - cs00[e]; partial += d * d;
        d = ct01[e] - cs01[e]; partial += d * d;
        d = ct10[e] - cs10[e]; partial += d * d;
        d = ct11[e] - cs11[e]; partial += d * d;
    }
    partial *= wgt;
    for (int off = 16; off > 0; off >>= 1)
        partial += __shfl_xor(partial, off, 32);

    if (lane == 0)
        atomicAdd(acc, partial);
}

// ---------------------------------------------------------------------------
// Scale partial sums by 1/(hw^2 * b) and emit (loss, [loss, p1, p2]).
// ---------------------------------------------------------------------------
__global__ void finalize_kernel(const float* __restrict__ acc,
                                float* __restrict__ out, float s1, float s2)
{
    if (blockIdx.x == 0 && threadIdx.x == 0) {
        float p1 = acc[0] * s1;
        float p2 = acc[1] * s2;
        float loss = p1 + p2;
        out[0] = loss;
        out[1] = loss;
        out[2] = p1;
        out[3] = p2;
    }
}

// ---------------------------------------------------------------------------
extern "C" void kernel_launch(void* const* d_in, const int* in_sizes, int n_in,
                              void* d_out, int out_size, void* d_ws, size_t ws_size,
                              hipStream_t stream)
{
    (void)in_sizes; (void)n_in; (void)out_size; (void)ws_size;

    const float* r1t = (const float*)d_in[0];   // [4,128,64,64]
    const float* r1s = (const float*)d_in[1];
    const float* r2t = (const float*)d_in[2];   // [4,256,32,32]
    const float* r2s = (const float*)d_in[3];
    float* out = (float*)d_out;

    const int B   = 4;
    const int C1  = 128, HW1 = 64 * 64;   // 4096
    const int C2  = 256, HW2 = 32 * 32;   // 1024

    char*  ws  = (char*)d_ws;
    float* acc = (float*)ws;              // 2 floats
    size_t off = 256;                     // keep panels 256B aligned
    _Float16* P1t = (_Float16*)(ws + off); off += (size_t)B * HW1 * C1 * sizeof(_Float16);
    _Float16* P1s = (_Float16*)(ws + off); off += (size_t)B * HW1 * C1 * sizeof(_Float16);
    _Float16* P2t = (_Float16*)(ws + off); off += (size_t)B * HW2 * C2 * sizeof(_Float16);
    _Float16* P2s = (_Float16*)(ws + off); // total ~12.6 MB

    init_acc_kernel<<<1, 32, 0, stream>>>(acc);

    const int tot1 = B * HW1;   // 16384 pixels
    const int tot2 = B * HW2;   // 4096 pixels
    normalize_kernel<<<(tot1 + 255) / 256, 256, 0, stream>>>(r1t, P1t, C1, HW1, tot1);
    normalize_kernel<<<(tot1 + 255) / 256, 256, 0, stream>>>(r1s, P1s, C1, HW1, tot1);
    normalize_kernel<<<(tot2 + 255) / 256, 256, 0, stream>>>(r2t, P2t, C2, HW2, tot2);
    normalize_kernel<<<(tot2 + 255) / 256, 256, 0, stream>>>(r2s, P2s, C2, HW2, tot2);

    // macro-tile triangle counts: n = hw/64, T = (n/2)*(n/2 + 1)
    const int n64_1 = HW1 / 64;                       // 64
    const int n64_2 = HW2 / 64;                       // 16
    const int T1 = (n64_1 / 2) * (n64_1 / 2 + 1);     // 1056
    const int T2 = (n64_2 / 2) * (n64_2 / 2 + 1);     // 72

    gram_diff_lds_kernel<128><<<dim3(T1, B), 256, 0, stream>>>(
        P1t, P1s, acc + 0, HW1, n64_1);
    gram_diff_lds_kernel<256><<<dim3(T2, B), 256, 0, stream>>>(
        P2t, P2s, acc + 1, HW2, n64_2);

    const float s1 = 1.0f / ((float)HW1 * (float)HW1 * (float)B);
    const float s2 = 1.0f / ((float)HW2 * (float)HW2 * (float)B);
    finalize_kernel<<<1, 32, 0, stream>>>(acc, out, s1, s2);
}
